// GNN_Sine_24086176596430
// MI455X (gfx1250) — matrix operand order repaired and checked by reading, not compiled, not run on hardware
//
#include <hip/hip_runtime.h>

typedef __attribute__((ext_vector_type(16))) _Float16 v16h;
typedef __attribute__((ext_vector_type(8)))  _Float16 v8h;
typedef __attribute__((ext_vector_type(8)))  float    v8f;

#define NJ 24
#define BT 16   // batch tile per block (= WMMA M)

// Tree adjacency (incl. self loops) as flat CSR neighbor lists, built from
// JOINT_TREES = [0,0,0,0,1,2,3,4,5,6,7,8,9,9,9,12,13,14,16,17,18,19,20,21]
__constant__ int c_nbr_off[NJ + 1] = {
    0, 4, 7, 10, 13, 16, 19, 22, 25, 28, 33, 35, 37,
    40, 43, 46, 48, 51, 54, 57, 60, 63, 66, 68, 70};
__constant__ int c_nbr[70] = {
    0, 1, 2, 3,
    1, 0, 4,
    2, 0, 5,
    3, 0, 6,
    4, 1, 7,
    5, 2, 8,
    6, 3, 9,
    7, 4, 10,
    8, 5, 11,
    9, 6, 12, 13, 14,
    10, 7,
    11, 8,
    12, 9, 15,
    13, 9, 16,
    14, 9, 17,
    15, 12,
    16, 13, 18,
    17, 14, 19,
    18, 16, 20,
    19, 17, 21,
    20, 18, 22,
    21, 19, 23,
    22, 20,
    23, 21};

// ---------------------------------------------------------------------------
// Weight pre-pack: fp32 [24][K][N] -> f16 WMMA B-fragments.
// Fragment (joint, kstep, ntile) = 32 lanes x 16 halves, contiguous, so the
// GEMM kernel loads B with one coalesced 32B-per-lane global load.
// B-matrix (32x16, f16) lane layout: lanes 0-15 hold N=lane, halves {0..7}=K 0..7,
// {8..15}=K 16..23; lanes 16-31 hold N=lane-16, halves {0..7}=K 8..15, {8..15}=K 24..31.
// ---------------------------------------------------------------------------
__global__ void pack_weights_kernel(const float* __restrict__ W,
                                    _Float16* __restrict__ dst,
                                    int K, int N, int ksteps, int ntiles) {
  int idx = blockIdx.x * blockDim.x + threadIdx.x;
  int total = NJ * ksteps * ntiles * 512;
  if (idx >= total) return;
  int half = idx & 15;
  int lane = (idx >> 4) & 31;
  int frag = idx >> 9;
  int ntile = frag % ntiles;
  int kstep = (frag / ntiles) % ksteps;
  int joint = frag / (ntiles * ksteps);
  int Kg = kstep * 32 + ((half < 8) ? (lane >> 4) * 8 + half : 16 + (lane >> 4) * 8 + (half - 8));
  int Ng = ntile * 16 + (lane & 15);
  float v = (Kg < K && Ng < N) ? W[((size_t)joint * K + Kg) * N + Ng] : 0.0f;
  dst[idx] = (_Float16)v;
}

// ---------------------------------------------------------------------------
// One fused GCN layer: per-joint WMMA linear -> LDS, then sparse weighted
// adjacency aggregation + bias (+ReLU) back into the activation buffer
// (or fp32 global output for the last layer).
// `wave` is passed through readfirstlane so all work-distribution control
// flow is scalar (s_cbranch, s_load) -- EXEC stays all-ones across WMMA.
// ---------------------------------------------------------------------------
template <int KSTEPS, int NTL, int NOUT, bool LAST>
__device__ __forceinline__ void gcn_layer(
    _Float16* __restrict__ act,          // LDS [24][16][64] f16
    _Float16* __restrict__ Hb,           // LDS [24][80][16] f16 (col-major)
    const _Float16* __restrict__ wfrag,  // packed B-fragments for this layer
    const float* __restrict__ aw,        // [24][24] learnable adjacency
    const float* __restrict__ bias,      // [NOUT]
    float* __restrict__ out, int baseB, int lane, int wave) {
  const int row16 = lane & 15;
  const int hiHalf = (lane >> 4) & 1;
  const int rowAdd = hiHalf * 8;

  // ---- per-joint linear: H[j] = act[j] (16xK) x W[j] (KxN), WMMA f16 ----
  for (int joint = wave; joint < NJ; joint += 8) {
    // prefetch the next joint's fragments this wave will need (L2-resident)
    if (joint + 8 < NJ)
      __builtin_prefetch(wfrag + (((joint + 8) * KSTEPS * NTL) << 9), 0, 1);

    // A-fragment per kstep, ISA 16-bit 16x32 layout:
    // lanes 0-15: M=lane, halves {0..7}=K 0..7,  {8..15}=K 16..23
    // lanes16-31: M=lane-16, halves {0..7}=K 8..15, {8..15}=K 24..31
    v16h afr[KSTEPS];
#pragma unroll
    for (int s = 0; s < KSTEPS; ++s) {
      const _Float16* ap =
          act + ((joint * 16 + row16) * 64 + s * 32 + hiHalf * 8);
      union { v16h v; v8h h[2]; } u;
      u.h[0] = *(const v8h*)ap;
      u.h[1] = *(const v8h*)(ap + 16);
      afr[s] = u.v;
    }
#pragma unroll
    for (int t = 0; t < NTL; ++t) {
      v8f acc = {};
#pragma unroll
      for (int s = 0; s < KSTEPS; ++s) {
        const v16h* bp = (const v16h*)(wfrag +
            (((joint * KSTEPS + s) * NTL + t) << 9));
        v16h bfr = bp[lane];
        acc = __builtin_amdgcn_wmma_f32_16x16x32_f16(
            false, afr[s], false, bfr, (short)0, acc, false, false);
      }
      // C layout: VGPR v -> row (v + rowAdd), col = lane&15 within tile.
      // Store column-major into Hb so each lane writes 8 contiguous halves.
      const int col = t * 16 + row16;
      v8h hv;
#pragma unroll
      for (int v = 0; v < 8; ++v) hv[v] = (_Float16)acc[v];
      *(v8h*)(Hb + ((joint * 80 + col) * 16 + rowAdd)) = hv;
    }
  }
  __syncthreads();

  // ---- sparse aggregation: out[n] = sum_m aw[n,m]*H[m] (+bias, ReLU) ----
  for (int task = wave; task < NJ * NTL; task += 8) {
    const int n = task / NTL;
    const int t = task - n * NTL;
    const int col = t * 16 + row16;
    float r[8];
#pragma unroll
    for (int v = 0; v < 8; ++v) r[v] = 0.0f;
    const int e0 = c_nbr_off[n], e1 = c_nbr_off[n + 1];
    for (int e = e0; e < e1; ++e) {
      const int m = c_nbr[e];
      const float w = aw[n * NJ + m];  // scalar (wave-uniform) -> s_load
      v8h hv = *(const v8h*)(Hb + ((m * 80 + col) * 16 + rowAdd));
#pragma unroll
      for (int v = 0; v < 8; ++v) r[v] = fmaf(w, (float)hv[v], r[v]);
    }
    if constexpr (!LAST) {
      const float bv = bias[col];
#pragma unroll
      for (int v = 0; v < 8; ++v) {
        float o = r[v] + bv;
        act[((n * 16 + rowAdd + v) * 64 + col)] = (_Float16)(o > 0.f ? o : 0.f);
      }
    } else {
      if (col < NOUT) {
        const float bv = bias[col];
#pragma unroll
        for (int v = 0; v < 8; ++v)
          out[((size_t)(baseB + rowAdd + v) * NJ + n) * NOUT + col] =
              r[v] + bv;
      }
    }
  }
  __syncthreads();
}

// ---------------------------------------------------------------------------
// Fused forward: 16 samples per block, all 4 layers, activations in LDS.
// ---------------------------------------------------------------------------
__global__ __launch_bounds__(256) void pngcn_fused_kernel(
    const float* __restrict__ x, const _Float16* __restrict__ wpack,
    const float* __restrict__ aw0, const float* __restrict__ b0,
    const float* __restrict__ aw1, const float* __restrict__ b1,
    const float* __restrict__ aw2, const float* __restrict__ b2,
    const float* __restrict__ aw3, const float* __restrict__ b3,
    float* __restrict__ out) {
  extern __shared__ char smem[];
  _Float16* act = (_Float16*)smem;      // [24][16][64] f16 = 49152 B
  _Float16* Hb = act + NJ * 16 * 64;    // [24][80][16] f16 = 61440 B

  const int tid = threadIdx.x;
  const int lane = tid & 31;
  // tid>>5 is wave-uniform; readfirstlane makes that visible to the compiler
  // so work-distribution loops become scalar control flow (no EXEC masking).
  const int wave = __builtin_amdgcn_readfirstlane(tid >> 5);
  const int baseB = blockIdx.x * BT;

  // zero activations (K-padding 13->32 and masked root joint stay 0)
  {
    v8h z = {};
    for (int i = tid; i < NJ * 16 * 64 / 8; i += 256) ((v8h*)act)[i] = z;
  }
  __syncthreads();

  // load x tile (mask_root: joint 0 left zero), convert fp32 -> f16
  for (int i = tid; i < BT * NJ * 13; i += 256) {
    int c = i % 13;
    int kb = i / 13;
    int k = kb % NJ;
    int b = kb / NJ;
    if (k != 0)
      act[(k * 16 + b) * 64 + c] =
          (_Float16)x[((size_t)(baseB + b) * NJ + k) * 13 + c];
  }
  __syncthreads();

  gcn_layer<1, 4, 64, false>(act, Hb, wpack + 0,      aw0, b0, out, baseB, lane, wave);
  gcn_layer<2, 4, 64, false>(act, Hb, wpack + 49152,  aw1, b1, out, baseB, lane, wave);
  gcn_layer<2, 4, 64, false>(act, Hb, wpack + 147456, aw2, b2, out, baseB, lane, wave);
  gcn_layer<2, 5, 65, true >(act, Hb, wpack + 245760, aw3, b3, out, baseB, lane, wave);
}

// ---------------------------------------------------------------------------
extern "C" void kernel_launch(void* const* d_in, const int* in_sizes, int n_in,
                              void* d_out, int out_size, void* d_ws,
                              size_t ws_size, hipStream_t stream) {
  const float* x   = (const float*)d_in[0];
  const float* w0  = (const float*)d_in[1];
  const float* aw0 = (const float*)d_in[2];
  const float* b0  = (const float*)d_in[3];
  const float* w1  = (const float*)d_in[4];
  const float* aw1 = (const float*)d_in[5];
  const float* b1  = (const float*)d_in[6];
  const float* w2  = (const float*)d_in[7];
  const float* aw2 = (const float*)d_in[8];
  const float* b2  = (const float*)d_in[9];
  const float* w3  = (const float*)d_in[10];
  const float* aw3 = (const float*)d_in[11];
  const float* b3  = (const float*)d_in[12];

  _Float16* wpack = (_Float16*)d_ws;  // needs 737280 B of scratch

  // pack weights into WMMA B-fragment order (f16)
  pack_weights_kernel<<<(49152 + 255) / 256, 256, 0, stream>>>(
      w0, wpack + 0, 13, 64, 1, 4);
  pack_weights_kernel<<<(98304 + 255) / 256, 256, 0, stream>>>(
      w1, wpack + 49152, 64, 64, 2, 4);
  pack_weights_kernel<<<(98304 + 255) / 256, 256, 0, stream>>>(
      w2, wpack + 147456, 64, 64, 2, 4);
  pack_weights_kernel<<<(122880 + 255) / 256, 256, 0, stream>>>(
      w3, wpack + 245760, 64, 65, 2, 5);

  const int B = in_sizes[0] / (NJ * 13);
  const int lds_bytes = (NJ * 16 * 64 + NJ * 80 * 16) * 2;  // 110592 B
  (void)hipFuncSetAttribute(
      reinterpret_cast<const void*>(pngcn_fused_kernel),
      hipFuncAttributeMaxDynamicSharedMemorySize, lds_bytes);
  pngcn_fused_kernel<<<B / BT, 256, lds_bytes, stream>>>(
      x, wpack, aw0, b0, aw1, b1, aw2, b2, aw3, b3, (float*)d_out);
}
